// Attention_42125039239602
// MI455X (gfx1250) — matrix operand order repaired and checked by reading, not compiled, not run on hardware
//
#include <hip/hip_runtime.h>
#include <hip/hip_bf16.h>

// ---------------------------------------------------------------------------
// Gated attention for MI455X (gfx1250, wave32, WMMA 16x16x32 f16 -> f32 acc)
//   x:[2,2048,512] f32, bias:[1,8,2048,2048] f32
//   out = (softmax(QK^T*scale + bias) @ V * sigmoid(xWg+bg)) @ Wo + bo
// Attention uses GLOBAL_LOAD_ASYNC_TO_LDS_B128 staging (ASYNCcnt) for K/V.
// ---------------------------------------------------------------------------

typedef __attribute__((ext_vector_type(16))) _Float16 v16h;
typedef __attribute__((ext_vector_type(8)))  _Float16 v8h;
typedef __attribute__((ext_vector_type(8)))  float    v8f;

#define HEADS   8
#define DHEAD   64
#define BATCH   2
#define SEQ     2048
#define DMODEL  512
#define ROWS    (BATCH * SEQ)           // 4096

// ---- WMMA wrapper ----------------------------------------------------------
__device__ __forceinline__ v8f wmma_f16(v16h a, v16h b, v8f c) {
    // D = A(16x32 f16) * B(32x16 f16) + C(16x16 f32)
    return __builtin_amdgcn_wmma_f32_16x16x32_f16(
        /*neg_a=*/false, a, /*neg_b=*/false, b,
        /*c_mod=*/(short)0, c, /*reuse_a=*/false, /*reuse_b=*/false);
}

// A-fragment (16x32, f16) from row-major A[M][ldA].
// Layout: lane<16 -> row=lane,    K=kb+0..15
//         lane>=16 -> row=lane-16, K=kb+16..31
__device__ __forceinline__ v16h load_a_frag(const _Float16* A, int ldA,
                                            int row_base, int kb, int lane) {
    int row  = row_base + (lane & 15);
    int koff = (lane < 16) ? 0 : 16;
    const _Float16* p = A + (size_t)row * ldA + kb + koff;
    return *(const v16h*)(p);
}

// B-fragment (32x16, f16) from Bt stored [N][ldB] (column n contiguous in K).
// Contiguous k order, same as A: lane<16 -> n=lane, K=kb+0..15 ; lane>=16 -> K=kb+16..31
__device__ __forceinline__ v16h load_b_frag(const _Float16* Bt, int ldB,
                                            int col_base, int kb, int lane) {
    int col = col_base + (lane & 15);
    int ks  = kb + ((lane < 16) ? 0 : 16);
    return *(const v16h*)(Bt + (size_t)col * ldB + ks);
}

// Async global->LDS copy of 16 bytes (CDNA5 TDM-class async path, ASYNCcnt).
// Low 32 bits of a generic pointer to __shared__ are the LDS byte offset
// (ISA 10.2).
__device__ __forceinline__ void async_copy_b128(const _Float16* gsrc,
                                                _Float16* ldst) {
    asm volatile("global_load_async_to_lds_b128 %0, %1, off"
                 :: "v"((unsigned int)(unsigned long long)ldst), "v"(gsrc)
                 : "memory");
}

// ---- prep kernels ----------------------------------------------------------
__global__ void cvt_f16_kernel(const float* __restrict__ in,
                               _Float16* __restrict__ out, int n) {
    int i = blockIdx.x * 256 + threadIdx.x;
    if (i < n) out[i] = (_Float16)in[i];
}

// in[K][N] f32 -> out[N][K] f16
__global__ void transpose_cvt_kernel(const float* __restrict__ in,
                                     _Float16* __restrict__ out,
                                     int K, int N) {
    int i = blockIdx.x * 256 + threadIdx.x;
    if (i < K * N) {
        int k = i / N, n = i % N;
        out[(size_t)n * K + k] = (_Float16)in[(size_t)k * N + n];
    }
}

// ---- fused QKV + gate projection ------------------------------------------
// Each wave: 16 rows x 64 cols of the concatenated [4096 x 2048] output
// (cols 0..511 = Q, 512..1535 = KV, 1536..2047 = gate).
__global__ __launch_bounds__(128) void proj_kernel(
    const _Float16* __restrict__ xh,    // [4096][512]
    const _Float16* __restrict__ Wqt,   // [512][512]  (N-major)
    const _Float16* __restrict__ Wkvt,  // [1024][512]
    const _Float16* __restrict__ Wgt,   // [512][512]
    const float*    __restrict__ bg,    // [512]
    _Float16* __restrict__ qh,          // [16][2048][64]  (scaled by dh^-0.5)
    _Float16* __restrict__ kh,          // [16][2048][64]
    _Float16* __restrict__ vT,          // [16][64][2048]
    float*    __restrict__ gate)        // [4096][512]
{
    int lane = threadIdx.x & 31;
    int wave = threadIdx.x >> 5;
    int m0 = blockIdx.x * 16;
    int c0 = (blockIdx.y * 4 + wave) * 64;

    const _Float16* Bt; int bcol0;
    if (c0 < 512)       { Bt = Wqt;  bcol0 = c0; }
    else if (c0 < 1536) { Bt = Wkvt; bcol0 = c0 - 512; }
    else                { Bt = Wgt;  bcol0 = c0 - 1536; }

    v8f acc[4] = {};
    for (int kb = 0; kb < DMODEL; kb += 32) {
        v16h a = load_a_frag(xh, DMODEL, m0, kb, lane);
#pragma unroll
        for (int t = 0; t < 4; ++t) {
            v16h b = load_b_frag(Bt, DMODEL, bcol0 + 16 * t, kb, lane);
            acc[t] = wmma_f16(a, b, acc[t]);
        }
    }

    int rowoff = (lane < 16) ? 0 : 8;
    int cl = lane & 15;
#pragma unroll
    for (int t = 0; t < 4; ++t) {
#pragma unroll
        for (int i = 0; i < 8; ++i) {
            int row = m0 + i + rowoff;
            int col = c0 + 16 * t + cl;
            float val = acc[t][i];
            int bb = row >> 11;          // /2048
            int nn = row & 2047;
            if (col < 512) {             // Q (scaled)
                int h = col >> 6, dh = col & 63;
                qh[(((size_t)(bb * HEADS + h)) * SEQ + nn) * DHEAD + dh] =
                    (_Float16)(val * 0.125f);
            } else if (col < 1024) {     // K
                int cc = col - 512; int h = cc >> 6, dh = cc & 63;
                kh[(((size_t)(bb * HEADS + h)) * SEQ + nn) * DHEAD + dh] =
                    (_Float16)val;
            } else if (col < 1536) {     // V (transposed: [dh][n])
                int cc = col - 1024; int h = cc >> 6, dh = cc & 63;
                vT[(((size_t)(bb * HEADS + h)) * DHEAD + dh) * SEQ + nn] =
                    (_Float16)val;
            } else {                     // gate = sigmoid(xWg + bg)
                int cc = col - 1536;
                float g = val + bg[cc];
                gate[(size_t)row * DMODEL + cc] = 1.0f / (1.0f + __expf(-g));
            }
        }
    }
}

// ---- flash attention -------------------------------------------------------
// Block = 4 waves sharing one head-batch; K/V tiles for each 32-key block are
// staged into LDS once per block via async global->LDS copies (ASYNCcnt),
// then consumed as WMMA B-fragments by all 4 waves (each owns a 16-row tile).
__global__ __launch_bounds__(128) void attn_kernel(
    const _Float16* __restrict__ qh,   // [16][2048][64]
    const _Float16* __restrict__ kh,   // [16][2048][64]
    const _Float16* __restrict__ vT,   // [16][64][2048]
    const float*    __restrict__ bias, // [1][8][2048][2048]
    const float*    __restrict__ gate, // [4096][512]
    _Float16*       __restrict__ ag)   // [4096][512] = (attn@V) * gate
{
    __shared__ _Float16 ksh[32 * 64];        // [key][dh]   4 KB
    __shared__ _Float16 vsh[64 * 32];        // [dh][key]   4 KB
    __shared__ _Float16 lds_p[4][16 * 32];   // per-wave P tile (transpose)

    int tid  = threadIdx.x;
    int lane = tid & 31;
    int wave = tid >> 5;
    int bh = blockIdx.y;
    int h  = bh & 7;
    int b  = bh >> 3;
    int r0 = (blockIdx.x * 4 + wave) * 16;

    const _Float16* q = qh + (size_t)bh * SEQ * DHEAD;
    const _Float16* k = kh + (size_t)bh * SEQ * DHEAD;
    const _Float16* v = vT + (size_t)bh * DHEAD * SEQ;
    const float* bptr = bias + (size_t)h * SEQ * SEQ;
    _Float16* pl = lds_p[wave];

    // static staging assignment: 128 threads move 8 KB (K 4KB + V 4KB)
    int kkey   = tid >> 2;            // 0..31
    int kchunk = (tid & 3) * 16;      // dh chunk: 0,16,32,48
    int vdh    = tid >> 1;            // 0..63
    int vko    = (tid & 1) * 16;      // key chunk: 0,16
    const _Float16* ksrc = k + (size_t)kkey * DHEAD + kchunk;  // + j0*DHEAD
    const _Float16* vsrc = v + (size_t)vdh * SEQ + vko;        // + j0
    _Float16* kdst = &ksh[kkey * DHEAD + kchunk];
    _Float16* vdst = &vsh[vdh * 32 + vko];

    // Q fragments (reused across all key blocks)
    v16h aq0 = load_a_frag(q, DHEAD, r0, 0,  lane);
    v16h aq1 = load_a_frag(q, DHEAD, r0, 32, lane);

    int rowoff = (lane < 16) ? 0 : 8;
    int cl = lane & 15;

    float m[8], s[8];
#pragma unroll
    for (int i = 0; i < 8; ++i) { m[i] = -1e30f; s[i] = 0.0f; }
    v8f acc[4] = {};

    for (int j0 = 0; j0 < SEQ; j0 += 32) {
        // ---- stage K[32x64] and V^T[64x32] tiles into LDS (async) ----
        __syncthreads();   // previous block's LDS consumers are done
        async_copy_b128(ksrc + (size_t)j0 * DHEAD,     kdst);
        async_copy_b128(ksrc + (size_t)j0 * DHEAD + 8, kdst + 8);
        async_copy_b128(vsrc + j0,                     vdst);
        async_copy_b128(vsrc + j0 + 8,                 vdst + 8);
        asm volatile("s_wait_asynccnt 0x0" ::: "memory");
        __syncthreads();   // tiles visible to all waves

        // ---- scores: two 16x16 tiles (keys j0..j0+31), bias added ----
        v8f S[2];
#pragma unroll
        for (int t = 0; t < 2; ++t) {
            v8f c = {};
            v16h bk0 = load_b_frag(ksh, DHEAD, 16 * t, 0,  lane);
            c = wmma_f16(aq0, bk0, c);
            v16h bk1 = load_b_frag(ksh, DHEAD, 16 * t, 32, lane);
            c = wmma_f16(aq1, bk1, c);
            const float* bp = bptr + (size_t)(r0 + rowoff) * SEQ
                                   + j0 + 16 * t + cl;
            __builtin_prefetch(bp + 32, 0, 0);   // next key block's bias
#pragma unroll
            for (int i = 0; i < 8; ++i)
                c[i] += bp[(size_t)i * SEQ];
            S[t] = c;
        }

        // ---- online softmax update (row reductions across 16-lane halves) --
        float p0[8], p1[8];
#pragma unroll
        for (int i = 0; i < 8; ++i) {
            float tm = fmaxf(S[0][i], S[1][i]);
            tm = fmaxf(tm, __shfl_xor(tm, 1, 16));
            tm = fmaxf(tm, __shfl_xor(tm, 2, 16));
            tm = fmaxf(tm, __shfl_xor(tm, 4, 16));
            tm = fmaxf(tm, __shfl_xor(tm, 8, 16));
            float nm = fmaxf(m[i], tm);
            float sc = __expf(m[i] - nm);
            m[i] = nm;
            float e0 = __expf(S[0][i] - nm);
            float e1 = __expf(S[1][i] - nm);
            p0[i] = e0; p1[i] = e1;
            float rs = e0 + e1;
            rs += __shfl_xor(rs, 1, 16);
            rs += __shfl_xor(rs, 2, 16);
            rs += __shfl_xor(rs, 4, 16);
            rs += __shfl_xor(rs, 8, 16);
            s[i] = s[i] * sc + rs;
#pragma unroll
            for (int t2 = 0; t2 < 4; ++t2) acc[t2][i] *= sc;
        }

        // ---- transpose P (C-layout -> A-layout) through per-wave LDS ----
#pragma unroll
        for (int i = 0; i < 8; ++i) {
            int row = i + rowoff;
            pl[row * 32 + cl]      = (_Float16)p0[i];
            pl[row * 32 + 16 + cl] = (_Float16)p1[i];
        }
        asm volatile("s_wait_dscnt 0x0" ::: "memory");
        v16h ap = load_a_frag(pl, 32, 0, 0, lane);

        // ---- O += P[16x32] @ V[32x64]  (V from LDS, [dh][key]) ----
#pragma unroll
        for (int t = 0; t < 4; ++t) {
            v16h bv = load_b_frag(vsh, 32, 16 * t, 0, lane);
            acc[t] = wmma_f16(ap, bv, acc[t]);
        }
    }

    // ---- normalize, gate, store f16 for the output GEMM ----
#pragma unroll
    for (int t = 0; t < 4; ++t) {
#pragma unroll
        for (int i = 0; i < 8; ++i) {
            int n  = r0 + i + rowoff;
            int dh = 16 * t + cl;
            size_t grow = (size_t)(b * SEQ + n);
            int gcol = h * DHEAD + dh;
            float o = acc[t][i] / s[i];
            float g = gate[grow * DMODEL + gcol];
            ag[grow * DMODEL + gcol] = (_Float16)(o * g);
        }
    }
}

// ---- output projection: out = ag @ Wo + bo --------------------------------
__global__ __launch_bounds__(128) void out_gemm_kernel(
    const _Float16* __restrict__ ag,   // [4096][512]
    const _Float16* __restrict__ Wot,  // [512][512] (N-major)
    const float*    __restrict__ bo,   // [512]
    float*          __restrict__ out)  // [4096][512]
{
    int lane = threadIdx.x & 31;
    int wave = threadIdx.x >> 5;
    int m0 = blockIdx.x * 16;
    int c0 = (blockIdx.y * 4 + wave) * 64;

    v8f acc[4] = {};
    for (int kb = 0; kb < DMODEL; kb += 32) {
        v16h a = load_a_frag(ag, DMODEL, m0, kb, lane);
#pragma unroll
        for (int t = 0; t < 4; ++t) {
            v16h b = load_b_frag(Wot, DMODEL, c0 + 16 * t, kb, lane);
            acc[t] = wmma_f16(a, b, acc[t]);
        }
    }
    int rowoff = (lane < 16) ? 0 : 8;
    int cl = lane & 15;
#pragma unroll
    for (int t = 0; t < 4; ++t) {
#pragma unroll
        for (int i = 0; i < 8; ++i) {
            int row = m0 + i + rowoff;
            int col = c0 + 16 * t + cl;
            out[(size_t)row * DMODEL + col] = acc[t][i] + bo[col];
        }
    }
}

// ---------------------------------------------------------------------------
extern "C" void kernel_launch(void* const* d_in, const int* in_sizes, int n_in,
                              void* d_out, int out_size, void* d_ws, size_t ws_size,
                              hipStream_t stream) {
    (void)in_sizes; (void)n_in; (void)out_size; (void)ws_size;

    const float* x    = (const float*)d_in[0];  // [2,2048,512]
    const float* bias = (const float*)d_in[1];  // [1,8,2048,2048]
    const float* Wq   = (const float*)d_in[2];  // [512,512]
    const float* Wkv  = (const float*)d_in[3];  // [512,1024]
    const float* Wo   = (const float*)d_in[4];  // [512,512]
    const float* bo   = (const float*)d_in[5];  // [512]
    const float* Wg   = (const float*)d_in[6];  // [512,512]
    const float* bg   = (const float*)d_in[7];  // [512]
    float* out = (float*)d_out;

    // workspace carve-up (all chunk sizes are multiples of 256 bytes)
    char* w = (char*)d_ws;
    _Float16* xh   = (_Float16*)w; w += (size_t)ROWS * DMODEL * 2;       // 4 MB
    _Float16* Wqt  = (_Float16*)w; w += (size_t)512 * 512 * 2;           // 0.5 MB
    _Float16* Wkvt = (_Float16*)w; w += (size_t)1024 * 512 * 2;          // 1 MB
    _Float16* Wgt  = (_Float16*)w; w += (size_t)512 * 512 * 2;           // 0.5 MB
    _Float16* Wot  = (_Float16*)w; w += (size_t)512 * 512 * 2;           // 0.5 MB
    _Float16* qh   = (_Float16*)w; w += (size_t)16 * SEQ * DHEAD * 2;    // 4 MB
    _Float16* kh   = (_Float16*)w; w += (size_t)16 * SEQ * DHEAD * 2;    // 4 MB
    _Float16* vT   = (_Float16*)w; w += (size_t)16 * DHEAD * SEQ * 2;    // 4 MB
    float*    gate = (float*)w;    w += (size_t)ROWS * DMODEL * 4;       // 8 MB
    _Float16* ag   = (_Float16*)w; w += (size_t)ROWS * DMODEL * 2;       // 4 MB

    // 1) conversions / weight transposes
    cvt_f16_kernel<<<(ROWS * DMODEL + 255) / 256, 256, 0, stream>>>(x, xh, ROWS * DMODEL);
    transpose_cvt_kernel<<<(512 * 512 + 255) / 256, 256, 0, stream>>>(Wq, Wqt, 512, 512);
    transpose_cvt_kernel<<<(512 * 1024 + 255) / 256, 256, 0, stream>>>(Wkv, Wkvt, 512, 1024);
    transpose_cvt_kernel<<<(512 * 512 + 255) / 256, 256, 0, stream>>>(Wg, Wgt, 512, 512);
    transpose_cvt_kernel<<<(512 * 512 + 255) / 256, 256, 0, stream>>>(Wo, Wot, 512, 512);

    // 2) fused Q/K/V/gate projections (WMMA)
    proj_kernel<<<dim3(ROWS / 16, 8), 128, 0, stream>>>(
        xh, Wqt, Wkvt, Wgt, bg, qh, kh, vT, gate);

    // 3) flash attention with async K/V LDS staging + gating epilogue (WMMA)
    attn_kernel<<<dim3(SEQ / 16 / 4, BATCH * HEADS), 128, 0, stream>>>(
        qh, kh, vT, bias, gate, ag);

    // 4) output projection + bo (WMMA)
    out_gemm_kernel<<<dim3(ROWS / 16, 2), 128, 0, stream>>>(ag, Wot, bo, out);
}
